// MichaelsRNN_88708254531768
// MI455X (gfx1250) — matrix operand order repaired and checked by reading, not compiled, not run on hardware
//
#include <hip/hip_runtime.h>
#include <hip/hip_bf16.h>

// ---------------------------------------------------------------------------
// MichaelsRNN on MI455X (gfx1250, wave32, WMMA).
// Batch rows are independent across the recurrence -> 16 persistent
// workgroups each own 32 batch rows for all T steps. ALL weights (J, I, fc_w
// as zero-padded bf16) plus fp32 state x and bf16 activations are resident in
// the WGP's 320KB LDS; per-step global traffic is only u[t] (prefetched) and
// the y stores. Matmuls: v_wmma_f32_16x16x32_bf16 (fp32 accum), register-
// tiled 2x in M so each J B-fragment feeds two WMMAs (halves LDS traffic per
// FLOP). Activation: hardware v_tanh_f32. LDS strides padded for banking.
// ---------------------------------------------------------------------------

typedef __attribute__((ext_vector_type(16))) __bf16        v16bf;
typedef __attribute__((ext_vector_type(8)))  float         v8f;
typedef __attribute__((ext_vector_type(4)))  unsigned int  v4u;
typedef __attribute__((ext_vector_type(8)))  unsigned int  v8u;

#define NN     300   // neurons
#define NPAD   304   // padded N (19 tiles of 16)
#define MT     32    // batch rows per workgroup (2 M-tiles)
#define BSZ    512
#define FIN    20
#define OD     50

// padded LDS row strides (elements); row starts stay 16B-aligned, strides
// chosen so 16 consecutive rows map to distinct LDS banks.
#define JSTR   328   // J:   [NPAD][JSTR] bf16, cols 0..319 used (10 K-tiles)
#define ASTR   328   // act: [MT][ASTR]  bf16, cols 0..319 used
#define ISTR   40    // I:   [NPAD][ISTR] bf16, cols 0..31 used
#define FSTR   136   // fcw: [64][FSTR]  bf16, cols 0..127 used
#define XSTR   308   // x:   [MT][XSTR]  f32,  cols 0..303 used
#define USTR   40    // u:   [MT][USTR]  bf16, cols 0..31 used

// float -> bf16, round-to-nearest-even
__device__ __forceinline__ unsigned short f2bf(float f) {
  unsigned int u = __builtin_bit_cast(unsigned int, f);
  u += 0x7FFFu + ((u >> 16) & 1u);
  return (unsigned short)(u >> 16);
}

// hardware tanh (gfx1250 TRANS op). Trailing v_nop satisfies the 1-instr
// TRANS result hazard (ISA 7.4) that the compiler cannot see through asm.
__device__ __forceinline__ float htanh(float x) {
  float r;
  asm volatile("v_tanh_f32 %0, %1\n\tv_nop" : "=v"(r) : "v"(x));
  return r;
}

// A-matrix fragment (16xK slice, bf16, row-major, stride in elements).
// Lane l (l<16): row M=l, K = {kbase..+7, kbase+16..+23}
// Lane l+16   : row M=l, K = {kbase+8..+15, kbase+24..+31}
__device__ __forceinline__ v16bf loadA(const unsigned short* base, int row,
                                       int stride, int kbase, int lane) {
  int k0 = kbase + ((lane >> 4) << 3);
  const unsigned short* p = base + row * stride + k0;
  v8u t;
  t.lo = *(const v4u*)(p);
  t.hi = *(const v4u*)(p + 16);
  return __builtin_bit_cast(v16bf, t);
}

// B-matrix fragment (Kx16 slice where B[k][n] = W[n][k], W row-major).
// Lane l (l<16): col N=l, K = {kbase..+7, kbase+16..+23} ; lane l+16: K = {kbase+8..+15, kbase+24..+31}
__device__ __forceinline__ v16bf loadB(const unsigned short* base, int row,
                                       int stride, int kbase, int lane) {
  int k0 = kbase + ((lane >> 4) << 3);
  const unsigned short* p = base + row * stride + k0;
  v8u t;
  t.lo = *(const v4u*)(p);
  t.hi = *(const v4u*)(p + 16);
  return __builtin_bit_cast(v16bf, t);
}

// ---------------------------------------------------------------------------
// Compile-time-specialized phases (CNT = N-tiles per wave: 3 or 2),
// register-tiled over 2 M-tiles so each B fragment feeds two WMMAs.
// ---------------------------------------------------------------------------
template <int CNT>
__device__ __forceinline__ void acc_phase(const unsigned short* acts,
                                          const unsigned short* us,
                                          const unsigned short* Jb,
                                          const unsigned short* Ib,
                                          int baseTile, int lane,
                                          v8f (&acc)[2][3]) {
  const int arow = lane & 15;
  const int bcol = lane & 15;
#pragma unroll 2
  for (int kt = 0; kt < 10; ++kt) {
    v16bf a0 = loadA(acts, arow,      ASTR, kt * 32, lane);
    v16bf a1 = loadA(acts, 16 + arow, ASTR, kt * 32, lane);
#pragma unroll
    for (int i = 0; i < CNT; ++i) {
      v16bf b = loadB(Jb, (baseTile + i) * 16 + bcol, JSTR, kt * 32, lane);
      acc[0][i] = __builtin_amdgcn_wmma_f32_16x16x32_bf16(
          false, a0, false, b, (short)0, acc[0][i], false, false);
      acc[1][i] = __builtin_amdgcn_wmma_f32_16x16x32_bf16(
          false, a1, false, b, (short)0, acc[1][i], false, false);
    }
  }
  // input projection: u @ I.T (single K-tile, F padded 20->32)
  v16bf a0 = loadA(us, arow,      USTR, 0, lane);
  v16bf a1 = loadA(us, 16 + arow, USTR, 0, lane);
#pragma unroll
  for (int i = 0; i < CNT; ++i) {
    v16bf b = loadB(Ib, (baseTile + i) * 16 + bcol, ISTR, 0, lane);
    acc[0][i] = __builtin_amdgcn_wmma_f32_16x16x32_bf16(
        false, a0, false, b, (short)0, acc[0][i], false, false);
    acc[1][i] = __builtin_amdgcn_wmma_f32_16x16x32_bf16(
        false, a1, false, b, (short)0, acc[1][i], false, false);
  }
}

template <int CNT>
__device__ __forceinline__ void write_phase(float* xs, unsigned short* acts,
                                            const float* __restrict__ Bvec,
                                            int baseTile, int lane,
                                            const v8f (&acc)[2][3]) {
#pragma unroll
  for (int mt = 0; mt < 2; ++mt) {
#pragma unroll
    for (int i = 0; i < CNT; ++i) {
      union { v8f v; float f[8]; } cu;
      cu.v = acc[mt][i];
      int n = (baseTile + i) * 16 + (lane & 15);
      float bv = (n < NN) ? Bvec[n] : 0.0f;
#pragma unroll
      for (int r = 0; r < 8; ++r) {
        int m = mt * 16 + r + ((lane >> 4) << 3);
        float xo  = xs[m * XSTR + n];
        float pre = 0.9f * xo + 0.1f * (cu.f[r] + bv);
        xs[m * XSTR + n]   = pre;
        acts[m * ASTR + n] = f2bf(fmaxf(htanh(pre), 0.0f));
      }
    }
  }
}

// ---------------------------------------------------------------------------
// Persistent recurrence kernel: 16 blocks x 256 threads (8 waves).
// LDS: J 199,424B + I 24,320B + fcw 17,408B + x 39,424B + act 20,992B +
//      u 2,560B = 304,128B of the WGP's 320KB.
// ---------------------------------------------------------------------------
__global__ __launch_bounds__(256) void rnn_kernel(
    const float* __restrict__ data, const float* __restrict__ Jg,
    const float* __restrict__ Ig, const float* __restrict__ Bvec,
    const float* __restrict__ x0, const float* __restrict__ fcwg,
    const float* __restrict__ fcb, float* __restrict__ out, int T) {
  __shared__ alignas(16) unsigned short Jb[NPAD * JSTR];
  __shared__ alignas(16) unsigned short Ib[NPAD * ISTR];
  __shared__ alignas(16) unsigned short Fb[64 * FSTR];
  __shared__ alignas(16) float          xs[MT * XSTR];
  __shared__ alignas(16) unsigned short acts[MT * ASTR];
  __shared__ alignas(16) unsigned short us[MT * USTR];

  const int tid  = threadIdx.x;
  const int lane = tid & 31;
  const int wave = tid >> 5;
  const int blockBase = blockIdx.x * MT;

  // N-tiles 0..8 -> waves 0..2 (3 each); tiles 9..18 -> waves 3..7 (2 each)
  const int baseTile = (wave < 3) ? wave * 3 : 9 + (wave - 3) * 2;

  // ---- one-time init: convert weights fp32 -> padded bf16 in LDS ----
  for (int idx = tid; idx < NPAD * JSTR; idx += 256) {
    int r = idx / JSTR, k = idx % JSTR;
    float v = (r < NN && k < NN) ? Jg[r * NN + k] : 0.0f;
    Jb[idx] = f2bf(v);
  }
  for (int idx = tid; idx < NPAD * ISTR; idx += 256) {
    int n = idx / ISTR, k = idx % ISTR;
    float v = (n < NN && k < FIN) ? Ig[n * FIN + k] : 0.0f;
    Ib[idx] = f2bf(v);
  }
  for (int idx = tid; idx < 64 * FSTR; idx += 256) {
    int o = idx / FSTR, k2 = idx % FSTR;
    int kg = 192 + k2;  // global act column
    float v = (o < OD && kg >= 200 && kg < 300) ? fcwg[o * 100 + (kg - 200)]
                                                : 0.0f;
    Fb[idx] = f2bf(v);
  }
  // ---- state init: x = tile(x0), act = retanh(x0), zero padding ----
  for (int idx = tid; idx < MT * XSTR; idx += 256) {
    int n = idx % XSTR;
    xs[idx] = (n < NN) ? x0[n] : 0.0f;
  }
  for (int idx = tid; idx < MT * ASTR; idx += 256) {
    int n = idx % ASTR;
    float a = (n < NN) ? fmaxf(htanh(x0[n]), 0.0f) : 0.0f;
    acts[idx] = f2bf(a);
  }
  __syncthreads();

  const int arow = lane & 15;

  for (int t = 0; t < T; ++t) {
    // ---- stage u[t] (pad F 20->32); prefetch u[t+1] into L2 ----
#pragma unroll
    for (int e = tid; e < MT * 32; e += 256) {
      int m = e >> 5, k = e & 31;
      float v = (k < FIN)
                    ? data[((size_t)t * BSZ + blockBase + m) * FIN + k]
                    : 0.0f;
      us[m * USTR + k] = f2bf(v);
    }
    if (t + 1 < T && tid < MT) {
      __builtin_prefetch(&data[((size_t)(t + 1) * BSZ + blockBase + tid) * FIN],
                         0, 0);
    }
    __syncthreads();

    // ---- accumulate: acc = act @ J.T + u @ I.T per 16x16 tile ----
    v8f acc[2][3];
    {
      v8f z = {};
      acc[0][0] = z; acc[0][1] = z; acc[0][2] = z;
      acc[1][0] = z; acc[1][1] = z; acc[1][2] = z;
    }
    if (wave < 3) {
      acc_phase<3>(acts, us, Jb, Ib, baseTile, lane, acc);
    } else {
      acc_phase<2>(acts, us, Jb, Ib, baseTile, lane, acc);
    }
    __syncthreads();  // all reads of acts done

    // ---- integrate + activation: pre = 0.9x + 0.1(acc + Bvec) ----
    if (wave < 3) {
      write_phase<3>(xs, acts, Bvec, baseTile, lane, acc);
    } else {
      write_phase<2>(xs, acts, Bvec, baseTile, lane, acc);
    }
    __syncthreads();  // acts updated

    // ---- readout: y = act[:, 200:300] @ fc_w.T + fc_b ----
    // exactly 8 tiles (2 M-tiles x 4 O-tiles) -> one per wave, no idle waves
    {
      const int rmt = wave >> 2;
      const int rot = wave & 3;
      v8f y = {};
#pragma unroll
      for (int kt = 0; kt < 4; ++kt) {
        v16bf a = loadA(acts, rmt * 16 + arow, ASTR, 192 + kt * 32, lane);
        v16bf b = loadB(Fb, rot * 16 + (lane & 15), FSTR, kt * 32, lane);
        y = __builtin_amdgcn_wmma_f32_16x16x32_bf16(
            false, a, false, b, (short)0, y, false, false);
      }
      union { v8f v; float f[8]; } yu;
      yu.v = y;
      int o = rot * 16 + (lane & 15);
      if (o < OD) {
        float fb = fcb[o];
#pragma unroll
        for (int r = 0; r < 8; ++r) {
          int m = rmt * 16 + r + ((lane >> 4) << 3);
          out[((size_t)t * BSZ + blockBase + m) * OD + o] = yu.f[r] + fb;
        }
      }
    }
    // next-iteration writes to us/acts are fenced by the barriers above.
  }
}

extern "C" void kernel_launch(void* const* d_in, const int* in_sizes, int n_in,
                              void* d_out, int out_size, void* d_ws,
                              size_t ws_size, hipStream_t stream) {
  const float* data = (const float*)d_in[0];  // [T,512,20]
  const float* J    = (const float*)d_in[1];  // [300,300]
  const float* I    = (const float*)d_in[2];  // [300,20]
  // d_in[3] = S : multiplied by hold==0, unused
  const float* Bvec = (const float*)d_in[4];  // [300,1]
  const float* x0   = (const float*)d_in[5];  // [300]
  const float* fcw  = (const float*)d_in[6];  // [50,100]
  const float* fcb  = (const float*)d_in[7];  // [50]
  float* out = (float*)d_out;

  const int T = in_sizes[0] / (BSZ * FIN);

  rnn_kernel<<<BSZ / MT, 256, 0, stream>>>(data, J, I, Bvec, x0, fcw, fcb,
                                           out, T);
}
